// EmbeddedLogRegClassifier_77558519431588
// MI455X (gfx1250) — matrix operand order repaired and checked by reading, not compiled, not run on hardware
//
#include <hip/hip_runtime.h>
#include <hip/hip_bf16.h>

typedef __attribute__((ext_vector_type(16))) _Float16 v16h;
typedef __attribute__((ext_vector_type(8)))  float    v8f;

#define B_SAMPLES 2048
#define V_VISITS  50
#define L_CODES   20
#define D_EMB     128
#define NCLASS    128
#define VL        (V_VISITS * L_CODES)   // 1000 indices per sample per table
#define K_DIM     (2 * D_EMB)            // 256

// ---------------------------------------------------------------------------
// Kernel 1: embedding-bag pooling.
// grid = (B, 2): blockIdx.y selects diag(0)/proc(1) table.
// 128 threads = 4 waves. Indices staged in LDS; each wave accumulates 250
// rows with coalesced float4 (b128) gathers; LDS cross-wave reduce; scale 1/L.
// Output: x[b, table*128 + j] in workspace, x is [2048, 256] row-major.
// ---------------------------------------------------------------------------
__global__ __launch_bounds__(128) void pool_kernel(
    const int* __restrict__ diag_idx,
    const int* __restrict__ proc_idx,
    const float* __restrict__ emb,
    float* __restrict__ x)
{
  __shared__ int    s_idx[VL];
  __shared__ float4 s_part[128];

  const int b     = blockIdx.x;
  const int table = blockIdx.y;
  const int tid   = threadIdx.x;
  const int lane  = tid & 31;
  const int wave  = tid >> 5;

  const int* __restrict__ idx =
      (table == 0 ? diag_idx : proc_idx) + (size_t)b * VL;

  for (int i = tid; i < VL; i += 128) s_idx[i] = idx[i];
  __syncthreads();

  float4 acc; acc.x = acc.y = acc.z = acc.w = 0.0f;
  // 1000 rows split 250/wave; 32 lanes * float4 = one 128-float row per step.
  for (int i = wave; i < VL; i += 4) {
    const float4* __restrict__ row =
        (const float4*)(emb + (size_t)s_idx[i] * D_EMB);
    float4 v = row[lane];
    acc.x += v.x; acc.y += v.y; acc.z += v.z; acc.w += v.w;
  }
  s_part[tid] = acc;
  __syncthreads();

  if (wave == 0) {
    float4 a0 = s_part[lane];
    float4 a1 = s_part[32 + lane];
    float4 a2 = s_part[64 + lane];
    float4 a3 = s_part[96 + lane];
    const float inv_l = 1.0f / (float)L_CODES;
    float4 t;
    t.x = (a0.x + a1.x + a2.x + a3.x) * inv_l;
    t.y = (a0.y + a1.y + a2.y + a3.y) * inv_l;
    t.z = (a0.z + a1.z + a2.z + a3.z) * inv_l;
    t.w = (a0.w + a1.w + a2.w + a3.w) * inv_l;
    float4* __restrict__ xout =
        (float4*)(x + (size_t)b * K_DIM + (size_t)table * D_EMB);
    xout[lane] = t;
  }
}

// ---------------------------------------------------------------------------
// Kernel 2: out[2048,128] = x[2048,256] @ W^T[256,128] + bias  via WMMA.
// 1024 tiles of 16x16; 8 waves/block, one tile per wave; K loop = 8 steps of
// 32 using v_wmma_f32_16x16x32_f16 (f32->f16 operand convert, f32 accum).
//
// Fragment layouts (CDNA5 ISA 7.12.2, wave32):
//  A (16x32 MxK): lane l: m = l%16, half = l/16;
//    elems 0..15 -> K = 16*half + e  (contiguous k)
//  B (32x16 KxN): lane l: n = l%16, half = l/16;
//    elems 0..15 -> K = 16*half + e        (same k order as A)
//  C/D (16x16 f32): lane l: n = l%16, half = l/16; VGPR r -> m = 8*half + r
// ---------------------------------------------------------------------------
__global__ __launch_bounds__(256) void gemm_wmma_kernel(
    const float* __restrict__ x,     // [2048, 256]
    const float* __restrict__ W,     // [128, 256]
    const float* __restrict__ bias,  // [128]
    float* __restrict__ out)         // [2048, 128]
{
  const int tid  = threadIdx.x;
  const int lane = tid & 31;
  const int wave = tid >> 5;
  const int tile = blockIdx.x * 8 + wave;   // 128 blocks * 8 waves = 1024 tiles
  const int m0   = (tile >> 3) * 16;        // 128 m-tiles
  const int n0   = (tile & 7) * 16;         // 8 n-tiles
  const int half = lane >> 4;
  const int sub  = lane & 15;

  v8f c = {};

  #pragma unroll
  for (int k0 = 0; k0 < K_DIM; k0 += 32) {
    v16h a, bm;

    // A fragment: row m0+sub of x
    const float* __restrict__ arow =
        x + (size_t)(m0 + sub) * K_DIM + k0 + half * 16;
    #pragma unroll
    for (int e = 0; e < 8; ++e)  a[e]     = (_Float16)arow[e];
    #pragma unroll
    for (int e = 0; e < 8; ++e)  a[8 + e] = (_Float16)arow[8 + e];

    // B fragment: row n0+sub of W (W[n,k] row-major; B(k,n)=W[n,k])
    const float* __restrict__ brow =
        W + (size_t)(n0 + sub) * K_DIM + k0 + half * 16;
    #pragma unroll
    for (int e = 0; e < 16; ++e) bm[e] = (_Float16)brow[e];

    c = __builtin_amdgcn_wmma_f32_16x16x32_f16(
        /*neg_a=*/false, a, /*neg_b=*/false, bm,
        /*c_mod=*/(short)0, c, /*reuse_a=*/false, /*reuse_b=*/false);
  }

  const float bn = bias[n0 + sub];
  #pragma unroll
  for (int r = 0; r < 8; ++r) {
    const int m = m0 + half * 8 + r;
    out[(size_t)m * NCLASS + n0 + sub] = c[r] + bn;
  }
}

// ---------------------------------------------------------------------------
extern "C" void kernel_launch(void* const* d_in, const int* in_sizes, int n_in,
                              void* d_out, int out_size, void* d_ws, size_t ws_size,
                              hipStream_t stream) {
  const int*   diag = (const int*)d_in[0];    // [B, V, L] int32
  const int*   proc = (const int*)d_in[1];    // [B, V, L] int32
  const float* emb  = (const float*)d_in[2];  // [VOCAB, 128] f32
  const float* W    = (const float*)d_in[3];  // [128, 256] f32
  const float* bias = (const float*)d_in[4];  // [128] f32
  float* out = (float*)d_out;                 // [2048, 128] f32
  float* x   = (float*)d_ws;                  // [2048, 256] f32 = 2 MB scratch

  pool_kernel<<<dim3(B_SAMPLES, 2), 128, 0, stream>>>(diag, proc, emb, x);
  gemm_wmma_kernel<<<dim3(128), 256, 0, stream>>>(x, W, bias, out);
}
